// ClassModule_23570780520589
// MI455X (gfx1250) — matrix operand order repaired and checked by reading, not compiled, not run on hardware
//
#include <hip/hip_runtime.h>

#define B_ 8
#define G_ 512
#define K_ 64
#define N_ 1024
#define M_ 128
#define E_ 4
#define L_ 5

typedef __attribute__((ext_vector_type(16))) _Float16 v16h;
typedef __attribute__((ext_vector_type(8)))  float    v8f;

__device__ __forceinline__ float wsum(float v) {
#pragma unroll
  for (int o = 16; o > 0; o >>= 1) v += __shfl_xor(v, o, 32);
  return v;
}
__device__ __forceinline__ float wmaxr(float v) {
#pragma unroll
  for (int o = 16; o > 0; o >>= 1) v = fmaxf(v, __shfl_xor(v, o, 32));
  return v;
}
__device__ __forceinline__ float4 exp4(float4 v) {
  return make_float4(__expf(v.x), __expf(v.y), __expf(v.z), __expf(v.w));
}

// pack 16 f32 (four float4) into a v16h operand register set
__device__ __forceinline__ v16h cvt16(float4 a, float4 b, float4 c, float4 d) {
  v16h v;
  v[0]  = (_Float16)a.x; v[1]  = (_Float16)a.y; v[2]  = (_Float16)a.z; v[3]  = (_Float16)a.w;
  v[4]  = (_Float16)b.x; v[5]  = (_Float16)b.y; v[6]  = (_Float16)b.z; v[7]  = (_Float16)b.w;
  v[8]  = (_Float16)c.x; v[9]  = (_Float16)c.y; v[10] = (_Float16)c.z; v[11] = (_Float16)c.w;
  v[12] = (_Float16)d.x; v[13] = (_Float16)d.y; v[14] = (_Float16)d.z; v[15] = (_Float16)d.w;
  return v;
}

// ---------------- streaming passes over z (HBM-bound; b128 loads, 1 wave/block) ---

// sumK[b,g,m] = sum_k exp(z[b,g,k,m]); one wave per (b,g), lane owns 4 m's
__global__ void k_exp_sums_K(const float* __restrict__ z, float* __restrict__ sumK) {
  int bg = blockIdx.x; int lane = threadIdx.x;
  const float4* zp = (const float4*)(z + (size_t)bg * K_ * M_) + lane;  // 32 f4 per row
  float4 s = make_float4(0.f, 0.f, 0.f, 0.f);
#pragma unroll 4
  for (int k = 0; k < K_; ++k) {
    float4 e = exp4(zp[(size_t)k * (M_ / 4)]);
    s.x += e.x; s.y += e.y; s.z += e.z; s.w += e.w;
  }
  ((float4*)sumK)[(size_t)bg * (M_ / 4) + lane] = s;
}

// sumG[b,k,m] = sum_g exp(z[b,g,k,m]); one wave per (b,k)
__global__ void k_exp_sums_G(const float* __restrict__ z, float* __restrict__ sumG) {
  int bk = blockIdx.x; int b = bk / K_; int k = bk % K_; int lane = threadIdx.x;
  const float4* zp = (const float4*)(z + ((size_t)b * G_ * K_ + k) * M_) + lane;
  float4 s = make_float4(0.f, 0.f, 0.f, 0.f);
#pragma unroll 4
  for (int g = 0; g < G_; ++g) {
    float4 e = exp4(zp[(size_t)g * (K_ * M_ / 4)]);
    s.x += e.x; s.y += e.y; s.z += e.z; s.w += e.w;
  }
  ((float4*)sumG)[(size_t)bk * (M_ / 4) + lane] = s;
}

// ra[b,g,m] = sum_k exp(z)/sumK * hs_key[b,k,m]; att[b,g,k] = z . z1_w + z1_b
// one wave per (b,g); lane owns 4 m's; att dot reduced across the wave
__global__ void k_ra_att(const float* __restrict__ z, const float* __restrict__ sumK,
                         const float* __restrict__ hs_key,
                         const float* __restrict__ z1_w, const float* __restrict__ z1_b,
                         float* __restrict__ ra, float* __restrict__ att) {
  int bg = blockIdx.x; int b = bg / G_; int lane = threadIdx.x;
  const float4* zp = (const float4*)(z + (size_t)bg * K_ * M_) + lane;
  const float4* hk = (const float4*)(hs_key + (size_t)b * K_ * M_) + lane;
  float4 sk = ((const float4*)sumK)[(size_t)bg * (M_ / 4) + lane];
  float4 inv = make_float4(1.f / sk.x, 1.f / sk.y, 1.f / sk.z, 1.f / sk.w);
  float4 zw = ((const float4*)z1_w)[lane];
  float zb = z1_b[0];
  float4 racc = make_float4(0.f, 0.f, 0.f, 0.f);
  for (int k = 0; k < K_; ++k) {
    float4 zv = zp[(size_t)k * (M_ / 4)];
    float4 h  = hk[(size_t)k * (M_ / 4)];
    float4 e  = exp4(zv);
    racc.x += e.x * inv.x * h.x; racc.y += e.y * inv.y * h.y;
    racc.z += e.z * inv.z * h.z; racc.w += e.w * inv.w * h.w;
    float t = zv.x * zw.x + zv.y * zw.y + zv.z * zw.z + zv.w * zw.w;
    t = wsum(t);
    if (lane == 0) att[(size_t)bg * K_ + k] = t + zb;
  }
  ((float4*)ra)[(size_t)bg * (M_ / 4) + lane] = racc;
}

// la[b,k,m] = sum_g exp(z)/sumG * hs_grid[b,g,m]; one wave per (b,k)
__global__ void k_la(const float* __restrict__ z, const float* __restrict__ sumG,
                     const float* __restrict__ hs_grid, float* __restrict__ la) {
  int bk = blockIdx.x; int b = bk / K_; int k = bk % K_; int lane = threadIdx.x;
  const float4* zp = (const float4*)(z + ((size_t)b * G_ * K_ + k) * M_) + lane;
  const float4* hg = (const float4*)(hs_grid + (size_t)b * G_ * M_) + lane;
  float4 sg = ((const float4*)sumG)[(size_t)bk * (M_ / 4) + lane];
  float4 inv = make_float4(1.f / sg.x, 1.f / sg.y, 1.f / sg.z, 1.f / sg.w);
  float4 acc = make_float4(0.f, 0.f, 0.f, 0.f);
#pragma unroll 2
  for (int g = 0; g < G_; ++g) {
    float4 e = exp4(zp[(size_t)g * (K_ * M_ / 4)]);
    float4 h = hg[(size_t)g * (M_ / 4)];
    acc.x += e.x * inv.x * h.x; acc.y += e.y * inv.y * h.y;
    acc.z += e.z * inv.z * h.z; acc.w += e.w * inv.w * h.w;
  }
  ((float4*)la)[(size_t)bk * (M_ / 4) + lane] = acc;
}

// ---------------- att softmaxes ----------------

// att_l[b,k] = sum_g softmax_K(att[b,g,:])[k]; block per b, 64 threads (k)
__global__ void k_att_l(const float* __restrict__ att, float* __restrict__ att_l) {
  int b = blockIdx.x; int k = threadIdx.x;
  __shared__ float red[2];
  float acc = 0.f;
  for (int g = 0; g < G_; ++g) {
    float x = att[((size_t)b * G_ + g) * K_ + k];
    float mx = wmaxr(x);
    if ((threadIdx.x & 31) == 0) red[threadIdx.x >> 5] = mx;
    __syncthreads();
    mx = fmaxf(red[0], red[1]);
    __syncthreads();
    float e = __expf(x - mx);
    float s = wsum(e);
    if ((threadIdx.x & 31) == 0) red[threadIdx.x >> 5] = s;
    __syncthreads();
    s = red[0] + red[1];
    __syncthreads();
    acc += e / s;
  }
  att_l[(size_t)b * K_ + k] = acc;
}

// att_r[b,g] = sum_k softmax_G(att[b,:,k])[g]; block per b, 512 threads (g)
__global__ void k_att_r(const float* __restrict__ att, float* __restrict__ att_r) {
  int b = blockIdx.x; int g = threadIdx.x;
  __shared__ float red[16];
  float acc = 0.f;
  for (int k = 0; k < K_; ++k) {
    float x = att[((size_t)b * G_ + g) * K_ + k];
    float mx = wmaxr(x);
    if ((threadIdx.x & 31) == 0) red[threadIdx.x >> 5] = mx;
    __syncthreads();
    mx = red[0];
#pragma unroll
    for (int i = 1; i < 16; ++i) mx = fmaxf(mx, red[i]);
    __syncthreads();
    float e = __expf(x - mx);
    float s = wsum(e);
    if ((threadIdx.x & 31) == 0) red[threadIdx.x >> 5] = s;
    __syncthreads();
    s = 0.f;
#pragma unroll
    for (int i = 0; i < 16; ++i) s += red[i];
    __syncthreads();
    acc += e / s;
  }
  att_r[(size_t)b * G_ + g] = acc;
}

// ---------------- generic strided batched WMMA GEMM (f32 in, f16 wmma, f32 out) ---
// D[b,r,c] = bias0[c]+bias1[c] + sum_k A0[b,r,k]*W0[b,c,k] (+ A1[b,r,k]*W1[b,c,k])
// CONTIG=true => ldAk==ldWk==1 and 16B-aligned rows: use float4 (b128) loads.
template <bool CONTIG>
__global__ void k_wmma_gemm(
    const float* __restrict__ A0, int ldAr, int ldAk, int bsA,
    const float* __restrict__ W0, int ldWn, int ldWk, int bsW,
    const float* __restrict__ A1, const float* __restrict__ W1,
    const float* __restrict__ bias0, const float* __restrict__ bias1,
    float* __restrict__ D, int ldDr, int bsD,
    int ctiles, int kin) {
  int rt = blockIdx.x / ctiles, ct = blockIdx.x % ctiles;
  int bb = blockIdx.z;
  const float* A  = A0 + (size_t)bb * bsA;
  const float* W  = W0 + (size_t)bb * bsW;
  const float* Ab = A1 ? A1 + (size_t)bb * bsA : nullptr;
  const float* Wb = W1 ? W1 + (size_t)bb * bsW : nullptr;
  int lane = threadIdx.x;
  int half = lane >> 4;          // lane half selects K sub-block per ISA layout
  int l15  = lane & 15;
  size_t arow = (size_t)(rt * 16 + l15) * ldAr;   // A row owned by this lane
  size_t wrow = (size_t)(ct * 16 + l15) * ldWn;   // B column owned by this lane
  v8f acc = {0.f, 0.f, 0.f, 0.f, 0.f, 0.f, 0.f, 0.f};
  for (int kk = 0; kk < kin; kk += 32) {
    // A 16x32 f16 layout: lanes<16 hold K {0..7,16..23}; lanes>=16 hold {8..15,24..31}
    int ka0 = kk + half * 8;
    int ka1 = kk + 16 + half * 8;
    // B 32x16 f16 layout: lanes<16 hold K {0..7,16..23}; lanes>=16 hold {8..15,24..31}
    int kb0 = kk + half * 8;
    v16h av, wv;
    if constexpr (CONTIG) {
      const float* Ap = A + arow;
      av = cvt16(*(const float4*)(Ap + ka0), *(const float4*)(Ap + ka0 + 4),
                 *(const float4*)(Ap + ka1), *(const float4*)(Ap + ka1 + 4));
      const float* Wp = W + wrow + kb0;
      wv = cvt16(*(const float4*)(Wp),     *(const float4*)(Wp + 4),
                 *(const float4*)(Wp + 16), *(const float4*)(Wp + 20));
    } else {
#pragma unroll
      for (int j = 0; j < 8; ++j) {
        av[j]     = (_Float16)A[arow + (size_t)(ka0 + j) * ldAk];
        av[8 + j] = (_Float16)A[arow + (size_t)(ka1 + j) * ldAk];
      }
#pragma unroll
      for (int j = 0; j < 16; ++j)
        wv[j] = (_Float16)W[wrow + (size_t)(kb0 + j + (j >= 8 ? 8 : 0)) * ldWk];
    }
    acc = __builtin_amdgcn_wmma_f32_16x16x32_f16(false, av, false, wv,
                                                 (short)0, acc, false, false);
    if (Ab) {
      v16h av2, wv2;
      if constexpr (CONTIG) {
        const float* Ap = Ab + arow;
        av2 = cvt16(*(const float4*)(Ap + ka0), *(const float4*)(Ap + ka0 + 4),
                    *(const float4*)(Ap + ka1), *(const float4*)(Ap + ka1 + 4));
        const float* Wp = Wb + wrow + kb0;
        wv2 = cvt16(*(const float4*)(Wp),     *(const float4*)(Wp + 4),
                    *(const float4*)(Wp + 16), *(const float4*)(Wp + 20));
      } else {
#pragma unroll
        for (int j = 0; j < 8; ++j) {
          av2[j]     = (_Float16)Ab[arow + (size_t)(ka0 + j) * ldAk];
          av2[8 + j] = (_Float16)Ab[arow + (size_t)(ka1 + j) * ldAk];
        }
#pragma unroll
        for (int j = 0; j < 16; ++j)
          wv2[j] = (_Float16)Wb[wrow + (size_t)(kb0 + j + (j >= 8 ? 8 : 0)) * ldWk];
      }
      acc = __builtin_amdgcn_wmma_f32_16x16x32_f16(false, av2, false, wv2,
                                                   (short)0, acc, false, false);
    }
  }
  int c = ct * 16 + l15;
  float badd = (bias0 ? bias0[c] : 0.f) + (bias1 ? bias1[c] : 0.f);
  float* Dp = D + (size_t)bb * bsD;
  // C/D layout: VGPR i -> row (i + half*8), col = l15
#pragma unroll
  for (int i = 0; i < 8; ++i) {
    int r = rt * 16 + half * 8 + i;
    Dp[(size_t)r * ldDr + c] = acc[i] + badd;
  }
}

// masked softmax over N (in place on zRl[b,n,k]); block per (b,k), 256 threads
__global__ void k_soft_N(float* __restrict__ zRl, const int* __restrict__ w_mask,
                         const float* __restrict__ w_Rl) {
  int b = blockIdx.x / K_; int k = blockIdx.x % K_;
  int t = threadIdx.x;
  __shared__ float red[8];
  float wrl = w_Rl[0];
  bool mz = (w_mask[(size_t)b * K_ + k] == 0);
  float* base = zRl + (size_t)b * N_ * K_ + k;
  float vals[4];
  float lm = -3.0e38f;
#pragma unroll
  for (int i = 0; i < 4; ++i) {
    int n = t + i * 256;
    float x = mz ? -1e10f : wrl * base[(size_t)n * K_];
    vals[i] = x;
    lm = fmaxf(lm, x);
  }
  lm = wmaxr(lm);
  if ((t & 31) == 0) red[t >> 5] = lm;
  __syncthreads();
  float mx = red[0];
#pragma unroll
  for (int i = 1; i < 8; ++i) mx = fmaxf(mx, red[i]);
  __syncthreads();
  float ls = 0.f;
#pragma unroll
  for (int i = 0; i < 4; ++i) { vals[i] = __expf(vals[i] - mx); ls += vals[i]; }
  ls = wsum(ls);
  if ((t & 31) == 0) red[t >> 5] = ls;
  __syncthreads();
  float s = 0.f;
#pragma unroll
  for (int i = 0; i < 8; ++i) s += red[i];
  float invs = 1.0f / s;
#pragma unroll
  for (int i = 0; i < 4; ++i)
    base[(size_t)(t + i * 256) * K_] = vals[i] * invs;
}

// final reductions + 260->5->1 head; block per b, 128 threads (m)
__global__ void k_final(const float* __restrict__ att_l, const float* __restrict__ att_r,
                        const float* __restrict__ la_lh, const float* __restrict__ kfrl,
                        const float* __restrict__ ra_rh, const float* __restrict__ lig_rep,
                        const float* __restrict__ mapL_w, const float* __restrict__ mapL_b,
                        const float* __restrict__ fin_w, const float* __restrict__ fin_b,
                        float* __restrict__ out) {
  int b = blockIdx.x; int m = threadIdx.x;
  __shared__ float pair[2 * M_ + E_];
  __shared__ float pl[L_];
  float kr = 0.f;
  for (int k = 0; k < K_; ++k) {
    float al = att_l[(size_t)b * K_ + k];
    size_t idx = ((size_t)b * K_ + k) * M_ + m;
    kr += al * (la_lh[idx] + kfrl[idx]);
  }
  float gr = 0.f;
  for (int g = 0; g < G_; ++g)
    gr += att_r[(size_t)b * G_ + g] * ra_rh[((size_t)b * G_ + g) * M_ + m];
  pair[m] = kr;                       // key_rep
  pair[M_ + E_ + m] = gr;             // grid_rep
  if (m < E_) pair[M_ + m] = lig_rep[(size_t)b * E_ + m];
  __syncthreads();
  if (m < L_) {
    float s = mapL_b[m];
    for (int i = 0; i < 2 * M_ + E_; ++i) s += mapL_w[m * (2 * M_ + E_) + i] * pair[i];
    pl[m] = s;
  }
  __syncthreads();
  if (m == 0) {
    float s = fin_b[0];
#pragma unroll
    for (int l = 0; l < L_; ++l) s += fin_w[l] * pl[l];
    out[b] = s;
  }
}

extern "C" void kernel_launch(void* const* d_in, const int* in_sizes, int n_in,
                              void* d_out, int out_size, void* d_ws, size_t ws_size,
                              hipStream_t stream) {
  (void)in_sizes; (void)n_in; (void)out_size; (void)ws_size;
  const float* z       = (const float*)d_in[0];
  const float* hs_grid = (const float*)d_in[1];
  const float* hs_key  = (const float*)d_in[2];
  const float* lig_rep = (const float*)d_in[3];
  const float* hs_rec  = (const float*)d_in[4];
  const float* w_Rl    = (const float*)d_in[5];
  const int*   w_mask  = (const int*)  d_in[6];
  const float* wra_w = (const float*)d_in[7],  *wra_b = (const float*)d_in[8];
  const float* wrh_w = (const float*)d_in[9],  *wrh_b = (const float*)d_in[10];
  const float* wla_w = (const float*)d_in[11], *wla_b = (const float*)d_in[12];
  const float* wlh_w = (const float*)d_in[13], *wlh_b = (const float*)d_in[14];
  const float* z1_w  = (const float*)d_in[15], *z1_b  = (const float*)d_in[16];
  const float* wcR_w = (const float*)d_in[17], *wcR_b = (const float*)d_in[18];
  const float* wcl_w = (const float*)d_in[19], *wcl_b = (const float*)d_in[20];
  const float* kR_w  = (const float*)d_in[21], *kR_b  = (const float*)d_in[22];
  const float* mapL_w = (const float*)d_in[23], *mapL_b = (const float*)d_in[24];
  const float* fin_w  = (const float*)d_in[25], *fin_b  = (const float*)d_in[26];
  float* out = (float*)d_out;

  float* ws = (float*)d_ws;
  size_t o = 0;
  float* sumK  = ws + o; o += (size_t)B_ * G_ * M_;
  float* sumG  = ws + o; o += (size_t)B_ * K_ * M_;
  float* ra    = ws + o; o += (size_t)B_ * G_ * M_;
  float* la    = ws + o; o += (size_t)B_ * K_ * M_;
  float* att   = ws + o; o += (size_t)B_ * G_ * K_;
  float* att_l = ws + o; o += (size_t)B_ * K_;
  float* att_r = ws + o; o += (size_t)B_ * G_;
  float* ra_rh = ws + o; o += (size_t)B_ * G_ * M_;
  float* la_lh = ws + o; o += (size_t)B_ * K_ * M_;
  float* Rh    = ws + o; o += (size_t)B_ * N_ * M_;
  float* lh    = ws + o; o += (size_t)B_ * K_ * M_;
  float* zRl   = ws + o; o += (size_t)B_ * N_ * K_;
  float* kfr   = ws + o; o += (size_t)B_ * K_ * M_;
  float* kfrl  = ws + o; o += (size_t)B_ * K_ * M_;

  // streaming passes over z (single-wave blocks, b128 per lane)
  k_exp_sums_K<<<B_ * G_, 32, 0, stream>>>(z, sumK);
  k_exp_sums_G<<<B_ * K_, 32, 0, stream>>>(z, sumG);
  k_ra_att<<<B_ * G_, 32, 0, stream>>>(z, sumK, hs_key, z1_w, z1_b, ra, att);
  k_la<<<B_ * K_, 32, 0, stream>>>(z, sumG, hs_grid, la);

  // att softmaxes
  k_att_l<<<B_, K_, 0, stream>>>(att, att_l);
  k_att_r<<<B_, G_, 0, stream>>>(att, att_r);

  const int CT_M = M_ / 16;   // 8
  // ra_rh = ra@wra^T + hs_grid@wrh^T + biases ; rows = B*G = 4096
  k_wmma_gemm<true><<<dim3((B_ * G_ / 16) * CT_M, 1, 1), 32, 0, stream>>>(
      ra, M_, 1, 0, wra_w, M_, 1, 0, hs_grid, wrh_w, wra_b, wrh_b,
      ra_rh, M_, 0, CT_M, M_);
  // la_lh = la@wla^T + hs_key@wlh^T + biases ; rows = B*K = 512
  k_wmma_gemm<true><<<dim3((B_ * K_ / 16) * CT_M, 1, 1), 32, 0, stream>>>(
      la, M_, 1, 0, wla_w, M_, 1, 0, hs_key, wlh_w, wla_b, wlh_b,
      la_lh, M_, 0, CT_M, M_);
  // Rh = hs_rec@wcR^T + b ; rows = B*N = 8192
  k_wmma_gemm<true><<<dim3((B_ * N_ / 16) * CT_M, 1, 1), 32, 0, stream>>>(
      hs_rec, M_, 1, 0, wcR_w, M_, 1, 0, nullptr, nullptr, wcR_b, nullptr,
      Rh, M_, 0, CT_M, M_);
  // lh = hs_key@wcl^T + b ; rows = B*K = 512
  k_wmma_gemm<true><<<dim3((B_ * K_ / 16) * CT_M, 1, 1), 32, 0, stream>>>(
      hs_key, M_, 1, 0, wcl_w, M_, 1, 0, nullptr, nullptr, wcl_b, nullptr,
      lh, M_, 0, CT_M, M_);
  // zRl[b] = Rh[b] (N x M) @ lh[b]^T (M x K) ; batched over b
  k_wmma_gemm<true><<<dim3((N_ / 16) * (K_ / 16), 1, B_), 32, 0, stream>>>(
      Rh, M_, 1, N_ * M_, lh, M_, 1, K_ * M_, nullptr, nullptr, nullptr, nullptr,
      zRl, K_, N_ * K_, K_ / 16, M_);
  // masked softmax over N (in place)
  k_soft_N<<<B_ * K_, 256, 0, stream>>>(zRl, w_mask, w_Rl);
  // kfr[b] = zRl[b]^T (K x N) @ hs_rec[b] (N x M); strided operands -> scalar path
  k_wmma_gemm<false><<<dim3((K_ / 16) * CT_M, 1, B_), 32, 0, stream>>>(
      zRl, 1, K_, N_ * K_, hs_rec, 1, M_, N_ * M_, nullptr, nullptr, nullptr, nullptr,
      kfr, M_, K_ * M_, CT_M, N_);
  // kfrl = kfr@kR^T + b ; rows = B*K = 512
  k_wmma_gemm<true><<<dim3((B_ * K_ / 16) * CT_M, 1, 1), 32, 0, stream>>>(
      kfr, M_, 1, 0, kR_w, M_, 1, 0, nullptr, nullptr, kR_b, nullptr,
      kfrl, M_, 0, CT_M, M_);

  // tail
  k_final<<<B_, M_, 0, stream>>>(att_l, att_r, la_lh, kfrl, ra_rh, lig_rep,
                                 mapL_w, mapL_b, fin_w, fin_b, out);
}
